// LNN_SNN_Block_56693568307790
// MI455X (gfx1250) — matrix operand order repaired and checked by reading, not compiled, not run on hardware
//
#include <hip/hip_runtime.h>

// ---------------------------------------------------------------------------
// Types for gfx1250 WMMA
// ---------------------------------------------------------------------------
typedef _Float16 v16h __attribute__((ext_vector_type(16)));
typedef _Float16 v8h  __attribute__((ext_vector_type(8)));
typedef float    v8f  __attribute__((ext_vector_type(8)));

union V16 { v16h v; v8h h[2]; };

#define B_      16
#define CIN_    32
#define COUT_   64
#define HW_     64
#define NPIX    65536      // B*H*W  (per channel)
#define NROWS   65536      // B*COUT*H  (rows of length 64 along W)
#define NOUT    4194304    // B*COUT*H*W

#define XP 72              // LDS pitch (halves) for the 16x64 staging tile
#define WP 72              // LDS pitch (halves) for the 64x64 weight tiles

// ---------------------------------------------------------------------------
// Utility kernels
// ---------------------------------------------------------------------------
__global__ void zero_kernel(float* __restrict__ p, int n) {
    int i = blockIdx.x * blockDim.x + threadIdx.x;
    if (i < n) p[i] = 0.0f;
}

// Fold BN: a = gamma*rsqrt(var+eps), c = beta - mean*a
__global__ void bn_finalize_kernel(const float* __restrict__ ssum,
                                   const float* __restrict__ ssq,
                                   const float* __restrict__ gamma,
                                   const float* __restrict__ beta,
                                   float* __restrict__ oscale,
                                   float* __restrict__ oshift) {
    int c = threadIdx.x;
    if (c < COUT_) {
        const float inv_n = 1.0f / (float)NPIX;
        float m  = ssum[c] * inv_n;
        float v  = ssq[c] * inv_n - m * m;
        float rs = rsqrtf(v + 1e-5f);
        float a  = rs * gamma[c];
        oscale[c] = a;
        oshift[c] = beta[c] - m * a;
    }
}

// ---------------------------------------------------------------------------
// Direct 3x3 conv (pad=1), NCHW. blockIdx.x = co, blockIdx.y = pixel chunk.
// Each thread: one output pixel. Block reduces sum/sumsq -> atomics.
// ---------------------------------------------------------------------------
__global__ __launch_bounds__(256)
void conv3x3_kernel(const float* __restrict__ x, const float* __restrict__ w,
                    const float* __restrict__ bias, float* __restrict__ y,
                    float* __restrict__ ssum, float* __restrict__ ssq, int cin) {
    __shared__ float wl[COUT_ * 9];        // up to 64*9 weights for this co
    __shared__ float r1[256];
    __shared__ float r2[256];

    const int tid = threadIdx.x;
    const int co  = blockIdx.x;

    for (int i = tid; i < cin * 9; i += 256)
        wl[i] = w[co * cin * 9 + i];
    __syncthreads();

    const int p  = blockIdx.y * 256 + tid;   // [0, 65536)
    const int b  = p >> 12;
    const int hw = p & 4095;
    const int h  = hw >> 6;
    const int wx = hw & 63;

    float acc = bias ? bias[co] : 0.0f;
    for (int ci = 0; ci < cin; ci++) {
        const float* xp = x + ((b * cin + ci) << 12);
        const float* wp = wl + ci * 9;
        #pragma unroll
        for (int kh = 0; kh < 3; kh++) {
            int ih = h + kh - 1;
            if (ih < 0 || ih > 63) continue;
            #pragma unroll
            for (int kw = 0; kw < 3; kw++) {
                int iw = wx + kw - 1;
                if (iw < 0 || iw > 63) continue;
                acc += xp[ih * 64 + iw] * wp[kh * 3 + kw];
            }
        }
    }
    y[((b * COUT_ + co) << 12) + hw] = acc;

    r1[tid] = acc;
    r2[tid] = acc * acc;
    __syncthreads();
    for (int s = 128; s > 0; s >>= 1) {
        if (tid < s) { r1[tid] += r1[tid + s]; r2[tid] += r2[tid + s]; }
        __syncthreads();
    }
    if (tid == 0) {
        atomicAdd(&ssum[co], r1[0]);
        atomicAdd(&ssq[co],  r2[0]);
    }
}

// 1x1 conv (shortcut), same stats scheme.
__global__ __launch_bounds__(256)
void conv1x1_kernel(const float* __restrict__ x, const float* __restrict__ w,
                    float* __restrict__ y,
                    float* __restrict__ ssum, float* __restrict__ ssq) {
    __shared__ float wl[CIN_];
    __shared__ float r1[256];
    __shared__ float r2[256];

    const int tid = threadIdx.x;
    const int co  = blockIdx.x;
    if (tid < CIN_) wl[tid] = w[co * CIN_ + tid];
    __syncthreads();

    const int p  = blockIdx.y * 256 + tid;
    const int b  = p >> 12;
    const int hw = p & 4095;

    float acc = 0.0f;
    #pragma unroll
    for (int ci = 0; ci < CIN_; ci++)
        acc += x[((b * CIN_ + ci) << 12) + hw] * wl[ci];
    y[((b * COUT_ + co) << 12) + hw] = acc;

    r1[tid] = acc;
    r2[tid] = acc * acc;
    __syncthreads();
    for (int s = 128; s > 0; s >>= 1) {
        if (tid < s) { r1[tid] += r1[tid + s]; r2[tid] += r2[tid + s]; }
        __syncthreads();
    }
    if (tid == 0) {
        atomicAdd(&ssum[co], r1[0]);
        atomicAdd(&ssq[co],  r2[0]);
    }
}

// ---------------------------------------------------------------------------
// Branch-free erf: Abramowitz-Stegun 7.1.26, max abs error ~1.5e-7.
// One v_rcp_f32 + one v_exp_f32 + FMAs, no EXEC divergence.
// ---------------------------------------------------------------------------
__device__ __forceinline__ float erf_fast(float x) {
    float ax = __builtin_fabsf(x);
    float t  = __builtin_amdgcn_rcpf(__builtin_fmaf(0.3275911f, ax, 1.0f));
    float p  = __builtin_fmaf(t, 1.061405429f, -1.453152027f);
    p = __builtin_fmaf(t, p, 1.421413741f);
    p = __builtin_fmaf(t, p, -0.284496736f);
    p = __builtin_fmaf(t, p, 0.254829592f);
    p = p * t;
    float e  = __expf(-ax * ax);
    float er = __builtin_fmaf(-p, e, 1.0f);
    return __builtin_copysignf(er, x);
}

// ---------------------------------------------------------------------------
// ODE + LIF kernel: the WMMA workhorse.
// Block = 128 threads = 4 wave32s; block owns 16 rows (one channel).
// Wave w owns output columns [16w, 16w+16). State kept in WMMA C/D layout.
// Each f(x): stage state (f16) to LDS -> read A (16x32 f16 layout) ->
// 4x v_wmma_f32_16x16x32_f16 (2 K-steps x {w_ode, tg_w^T}, shared A) ->
// branch-free gelu/sigmoid elementwise. Weights pre-staged to LDS once,
// B-tiles hoisted into registers (invariant across all 30 f-evals).
// ---------------------------------------------------------------------------
__global__ __launch_bounds__(128)
void ode_lif_kernel(const float* __restrict__ y,
                    const float* __restrict__ bnscale,
                    const float* __restrict__ bnshift,
                    const float* __restrict__ wode,
                    const float* __restrict__ tgw,
                    const float* __restrict__ tgb,
                    float thr,
                    float* __restrict__ out) {
    __shared__ __align__(16) _Float16 lds_wt[COUT_ * WP];  // wode transposed
    __shared__ __align__(16) _Float16 lds_tg[COUT_ * WP];  // tg_w row-major
    __shared__ __align__(16) _Float16 lds_x[16 * XP];      // A staging tile

    const int tid     = threadIdx.x;
    const int wv      = tid >> 5;          // wave id 0..3 (wave32)
    const int lane    = tid & 31;
    const int rowBase = blockIdx.x * 16;
    const int ch      = (rowBase >> 6) & 63;

    // Stage weights as f16. U = x @ wode  => B[k][n] = wode[k][n] (transpose
    // into n-major rows). V = x @ tg^T    => B[k][j] = tg[j][k]  (row-major).
    for (int i = tid; i < COUT_ * COUT_; i += 128) {
        int n = i >> 6, k = i & 63;
        lds_wt[n * WP + k] = (_Float16)wode[k * 64 + n];
        lds_tg[n * WP + k] = (_Float16)tgw[n * 64 + k];
    }
    __syncthreads();

    // Hoist this wave's B tiles into registers (invariant).
    // B 32x16 f16 layout: lane (N = lane&15) holds K [bko..bko+7] and [bko+16..bko+23], bko = (lane>=16 ? 8 : 0).
    const int bcol = 16 * wv + (lane & 15);
    const int bko  = (lane & 16) ? 8 : 0;
    V16 BU0, BU1, BT0, BT1;
    BU0.h[0] = *(const v8h*)&lds_wt[bcol * WP + 0  + bko];
    BU0.h[1] = *(const v8h*)&lds_wt[bcol * WP + 0  + bko + 16];
    BU1.h[0] = *(const v8h*)&lds_wt[bcol * WP + 32 + bko];
    BU1.h[1] = *(const v8h*)&lds_wt[bcol * WP + 32 + bko + 16];
    BT0.h[0] = *(const v8h*)&lds_tg[bcol * WP + 0  + bko];
    BT0.h[1] = *(const v8h*)&lds_tg[bcol * WP + 0  + bko + 16];
    BT1.h[0] = *(const v8h*)&lds_tg[bcol * WP + 32 + bko];
    BT1.h[1] = *(const v8h*)&lds_tg[bcol * WP + 32 + bko + 16];

    // C/D 16x16 f32 layout: lane holds (M = j + (lane>=16 ? 8:0), N = lane&15 + 16*wv).
    const int col  = 16 * wv + (lane & 15);
    const int rb   = (lane & 16) ? 8 : 0;
    const float tgbv = tgb[col];

    // Load input rows + apply folded BN affine.
    const float a  = bnscale[ch];
    const float sh = bnshift[ch];
    v8f x;
    #pragma unroll
    for (int j = 0; j < 8; j++)
        x[j] = a * y[(rowBase + rb + j) * 64 + col] + sh;

    auto feval = [&](v8f xs) -> v8f {
        __syncthreads();   // protect previous iteration's A readers
        #pragma unroll
        for (int j = 0; j < 8; j++)
            lds_x[(rb + j) * XP + col] = (_Float16)xs[j];
        __syncthreads();

        // A 16x32 f16 layout: lane (M = lane&15) reads halves
        // [ak..ak+7] and [ak+16..ak+23] of its row, ak = (lane>=16 ? 8 : 0).
        const int arow = lane & 15;
        const int ak   = (lane & 16) ? 8 : 0;
        V16 A0, A1;
        A0.h[0] = *(const v8h*)&lds_x[arow * XP + ak];
        A0.h[1] = *(const v8h*)&lds_x[arow * XP + ak + 16];
        A1.h[0] = *(const v8h*)&lds_x[arow * XP + 32 + ak];
        A1.h[1] = *(const v8h*)&lds_x[arow * XP + 32 + ak + 16];

        v8f U = {0.f,0.f,0.f,0.f,0.f,0.f,0.f,0.f};
        v8f V = {0.f,0.f,0.f,0.f,0.f,0.f,0.f,0.f};
        U = __builtin_amdgcn_wmma_f32_16x16x32_f16(false, A0.v, false, BU0.v, (short)0, U, false, false);
        U = __builtin_amdgcn_wmma_f32_16x16x32_f16(false, A1.v, false, BU1.v, (short)0, U, false, false);
        V = __builtin_amdgcn_wmma_f32_16x16x32_f16(false, A0.v, false, BT0.v, (short)0, V, false, false);
        V = __builtin_amdgcn_wmma_f32_16x16x32_f16(false, A1.v, false, BT1.v, (short)0, V, false, false);

        v8f k;
        #pragma unroll
        for (int j = 0; j < 8; j++) {
            float u = U[j];
            // exact-form gelu via branch-free erf approximation
            float g = 0.5f * u * (1.0f + erf_fast(u * 0.70710678118654752f));
            // tf = 1/(1+sigmoid(z)) = (1+e^-z)/(2+e^-z)  (one exp, one rcp)
            float z  = V[j] + tgbv;
            float ez = __expf(-z);
            float tf = (1.0f + ez) * __builtin_amdgcn_rcpf(2.0f + ez);
            k[j] = g * tf;
        }
        return k;
    };

    // dopri5, fixed h = 1/5, 5 steps.
    const float H   = 0.2f;
    const float A21 = 0.2f;
    const float A31 = 0.075f, A32 = 0.225f;
    const float A41 = 0.97777777777777777f, A42 = -3.7333333333333334f, A43 = 3.5555555555555554f;
    const float A51 = 2.9525986892242035f,  A52 = -11.595793324188385f,
                A53 = 9.8228928516994358f,  A54 = -0.29080932784636487f;
    const float A61 = 2.8462752525252526f,  A62 = -10.757575757575758f,
                A63 = 8.9064227177434727f,  A64 = 0.27840909090909091f, A65 = -0.27353130360205832f;
    const float BB1 = 0.091145833333333333f, BB3 = 0.44923629829290207f,
                BB4 = 0.65104166666666667f,  BB5 = -0.32237617924528302f, BB6 = 0.13095238095238096f;

    #pragma unroll 1
    for (int stp = 0; stp < 5; stp++) {
        v8f k1 = feval(x);
        v8f k2 = feval(x + (H * A21) * k1);
        v8f k3 = feval(x + H * (A31 * k1 + A32 * k2));
        v8f k4 = feval(x + H * (A41 * k1 + A42 * k2 + A43 * k3));
        v8f k5 = feval(x + H * (A51 * k1 + A52 * k2 + A53 * k3 + A54 * k4));
        v8f k6 = feval(x + H * (A61 * k1 + A62 * k2 + A63 * k3 + A64 * k4 + A65 * k5));
        x = x + H * (BB1 * k1 + BB3 * k3 + BB4 * k4 + BB5 * k5 + BB6 * k6);
    }

    // spike forward: (x - thr) > 0
    #pragma unroll
    for (int j = 0; j < 8; j++)
        out[(rowBase + rb + j) * 64 + col] = (x[j] > thr) ? 1.0f : 0.0f;
}

// ---------------------------------------------------------------------------
// out += BN(shortcut)
// ---------------------------------------------------------------------------
__global__ __launch_bounds__(256)
void add_residual_kernel(float* __restrict__ out, const float* __restrict__ resid,
                         const float* __restrict__ a, const float* __restrict__ b) {
    int i = blockIdx.x * 256 + threadIdx.x;
    if (i < NOUT) {
        int c = (i >> 12) & 63;
        out[i] = out[i] + a[c] * resid[i] + b[c];
    }
}

// ---------------------------------------------------------------------------
// Launch
// ---------------------------------------------------------------------------
extern "C" void kernel_launch(void* const* d_in, const int* in_sizes, int n_in,
                              void* d_out, int out_size, void* d_ws, size_t ws_size,
                              hipStream_t stream) {
    const float* x       = (const float*)d_in[0];
    const float* conv1_w = (const float*)d_in[1];
    const float* conv1_b = (const float*)d_in[2];
    const float* bn1_g   = (const float*)d_in[3];
    const float* bn1_b   = (const float*)d_in[4];
    const float* ode1_w  = (const float*)d_in[5];
    const float* tg1_w   = (const float*)d_in[6];
    const float* tg1_b   = (const float*)d_in[7];
    const float* conv2_w = (const float*)d_in[8];
    const float* bn2_g   = (const float*)d_in[9];
    const float* bn2_b   = (const float*)d_in[10];
    const float* ode2_w  = (const float*)d_in[11];
    const float* tg2_w   = (const float*)d_in[12];
    const float* tg2_b   = (const float*)d_in[13];
    const float* sc_w    = (const float*)d_in[14];
    const float* sc_g    = (const float*)d_in[15];
    const float* sc_b    = (const float*)d_in[16];

    float* bufA = (float*)d_ws;            // conv outputs (y1, then y2)
    float* bufB = bufA + NOUT;             // h1 spikes, then shortcut conv
    float* st   = bufB + NOUT;             // stats
    float* ssum = st;                      // 3*64
    float* ssq  = st + 192;                // 3*64
    float* sca  = st + 384;                // 3*64 folded BN scale
    float* shf  = st + 576;                // 3*64 folded BN shift

    float* outf = (float*)d_out;

    // 0) zero stats accumulators (deterministic every call)
    zero_kernel<<<2, 256, 0, stream>>>(st, 384);

    // 1) conv1 + stats
    conv3x3_kernel<<<dim3(COUT_, 256), 256, 0, stream>>>(x, conv1_w, conv1_b,
                                                         bufA, ssum, ssq, CIN_);
    bn_finalize_kernel<<<1, 64, 0, stream>>>(ssum, ssq, bn1_g, bn1_b, sca, shf);

    // 2) LIF layer 1: BN fold + dopri5 ODE (WMMA) + spike(0.3) -> bufB
    ode_lif_kernel<<<NROWS / 16, 128, 0, stream>>>(bufA, sca, shf,
                                                   ode1_w, tg1_w, tg1_b, 0.3f, bufB);

    // 3) conv2 + stats
    conv3x3_kernel<<<dim3(COUT_, 256), 256, 0, stream>>>(bufB, conv2_w, nullptr,
                                                         bufA, ssum + 64, ssq + 64, COUT_);
    bn_finalize_kernel<<<1, 64, 0, stream>>>(ssum + 64, ssq + 64, bn2_g, bn2_b,
                                             sca + 64, shf + 64);

    // 4) LIF layer 2 -> spikes written straight into d_out
    ode_lif_kernel<<<NROWS / 16, 128, 0, stream>>>(bufA, sca + 64, shf + 64,
                                                   ode2_w, tg2_w, tg2_b, 0.5f, outf);

    // 5) shortcut 1x1 conv + stats (bufB free after conv2 consumed it)
    conv1x1_kernel<<<dim3(COUT_, 256), 256, 0, stream>>>(x, sc_w, bufB,
                                                         ssum + 128, ssq + 128);
    bn_finalize_kernel<<<1, 64, 0, stream>>>(ssum + 128, ssq + 128, sc_g, sc_b,
                                             sca + 128, shf + 128);

    // 6) out = spikes + BN(shortcut)
    add_residual_kernel<<<NOUT / 256, 256, 0, stream>>>(outf, bufB, sca + 128, shf + 128);
}
